// SelfAttention_v1_89129161326937
// MI455X (gfx1250) — compile-verified, matrix operand order checked by reading, not run on hardware
//
#include <hip/hip_runtime.h>

// Self-attention N=8192, D_IN=256, D_OUT=128 on gfx1250 (wave32 WMMA).
// Pipeline:
//   k0: x fp32 -> f16
//   k1: W{q,k,v} fp32 [256,128] -> f16 transposed [128,256]
//   k2: WMMA GEMM -> Qh (scaled, [8192,128] f16), Kh ([8192,128] f16),
//       Vt ([128,8192] f16, transposed for P*V B-operand)
//   k3: split-K flash attention (4 key segments, online softmax partials)
//   k4: combine partials -> f32 out [8192,128]

#define NTOK  8192
#define DIN   256
#define DH    128
#define NSEG  4                      // key-dimension split for occupancy
#define SEGK  (NTOK / NSEG)          // 2048 keys per segment

typedef __attribute__((ext_vector_type(16))) _Float16 v16h;
typedef __attribute__((ext_vector_type(8)))  _Float16 v8h;
typedef __attribute__((ext_vector_type(8)))  float    v8f;

union Frag16 { v16h v; v8h h[2]; };

// A/B 16-bit fragment load (ISA 7.12.2): lane holds row (A) / column (B^T),
// 16 halves = chunk [kbase+8*hi .. +7] and [kbase+16+8*hi .. +7].
__device__ __forceinline__ v16h load_frag(const _Float16* p, int hi) {
    const _Float16* q = p + (hi << 3);
    Frag16 f;
    f.h[0] = *(const v8h*)(q);
    f.h[1] = *(const v8h*)(q + 16);
    return f.v;
}

__device__ __forceinline__ float red_max16(float v) {
#pragma unroll
    for (int m = 8; m; m >>= 1) v = fmaxf(v, __shfl_xor(v, m, 32));
    return v;
}
__device__ __forceinline__ float red_sum16(float v) {
#pragma unroll
    for (int m = 8; m; m >>= 1) v += __shfl_xor(v, m, 32);
    return v;
}

// ---------------- k0: convert x to f16 ----------------
__global__ void cvt_x_kernel(const float* __restrict__ x,
                             _Float16* __restrict__ xh, int n) {
    int i = blockIdx.x * blockDim.x + threadIdx.x;
    if (i < n) xh[i] = (_Float16)x[i];
}

// ---------------- k1: convert + transpose weights ----------------
// wt layout: [3][128][256] f16; wt[mat][n][k] = W[mat][k][n]
__global__ void cvt_w_kernel(const float* __restrict__ wq,
                             const float* __restrict__ wk,
                             const float* __restrict__ wv,
                             _Float16* __restrict__ wt) {
    int i = blockIdx.x * blockDim.x + threadIdx.x;
    if (i >= 3 * DH * DIN) return;
    int mat = i >> 15;            // 128*256 = 32768 per matrix
    int rem = i & 32767;
    int n = rem >> 8;             // 0..127
    int k = rem & 255;            // 0..255
    const float* w = (mat == 0) ? wq : (mat == 1) ? wk : wv;
    wt[i] = (_Float16)w[k * DH + n];
}

// ---------------- k2: QKV projection GEMM (WMMA) ----------------
// One wave computes one 16x16 tile. 3 matrices * (512 Mtiles * 8 Ntiles).
__global__ __launch_bounds__(256) void qkv_gemm_kernel(
    const _Float16* __restrict__ xh,   // [8192,256]
    const _Float16* __restrict__ wt,   // [3][128][256]
    _Float16* __restrict__ Qh,         // [8192,128] (scaled)
    _Float16* __restrict__ Kh,         // [8192,128]
    _Float16* __restrict__ Vt) {       // [128,8192]
    int wave = (blockIdx.x * blockDim.x + threadIdx.x) >> 5;
    int lane = threadIdx.x & 31;
    int mat  = wave >> 12;             // 4096 tiles per matrix
    int tile = wave & 4095;
    int mt = tile >> 3;                // 0..511
    int nt = tile & 7;                 // 0..7
    int m  = lane & 15;
    int hi = lane >> 4;

    const _Float16* arow = xh + (size_t)(mt * 16 + m) * DIN;
    const _Float16* brow = wt + (size_t)mat * DH * DIN + (size_t)(nt * 16 + m) * DIN;

    v8f c = {};
#pragma unroll
    for (int kk = 0; kk < DIN / 32; ++kk) {
        v16h a = load_frag(arow + kk * 32, hi);
        v16h b = load_frag(brow + kk * 32, hi);
        c = __builtin_amdgcn_wmma_f32_16x16x32_f16(false, a, false, b,
                                                   (short)0, c, false, false);
    }
    const float scale = 0.08838834764831845f;   // 1/sqrt(128)
#pragma unroll
    for (int r = 0; r < 8; ++r) {
        int gm = mt * 16 + r + 8 * hi;          // C layout: row r+8*hi
        int gn = nt * 16 + m;                   // col = lane&15
        float v = c[r];
        if (mat == 0)      Qh[(size_t)gm * DH + gn] = (_Float16)(v * scale);
        else if (mat == 1) Kh[(size_t)gm * DH + gn] = (_Float16)v;
        else               Vt[(size_t)gn * NTOK + gm] = (_Float16)v;
    }
}

// ---------------- k3: split-K flash attention ----------------
// wave = (seg, row-tile): each wave owns 16 Q rows over a 2048-key segment.
// Emits unnormalized partials: O_s = sum exp(s-m_s)*V, m_s, l_s.
__global__ __launch_bounds__(128) void flash_attn_partial_kernel(
    const _Float16* __restrict__ Qh,   // [8192,128], pre-scaled
    const _Float16* __restrict__ Kh,   // [8192,128]
    const _Float16* __restrict__ Vt,   // [128,8192]
    float* __restrict__ Opart,         // [NSEG][8192][128]
    float* __restrict__ Mpart,         // [NSEG][8192]
    float* __restrict__ Lpart) {       // [NSEG][8192]
    __shared__ alignas(16) _Float16 pbuf[4][16 * 32];   // per-wave P staging

    int wiw  = threadIdx.x >> 5;
    int lane = threadIdx.x & 31;
    int wave = blockIdx.x * 4 + wiw;   // 0..2047
    int tile = wave & 511;             // Q row tile
    int seg  = wave >> 9;              // key segment
    int m  = lane & 15;
    int hi = lane >> 4;
    int m0 = tile * 16;
    int kstart = seg * SEGK;
    int kend   = kstart + SEGK;
    _Float16* pl = &pbuf[wiw][0];

    // Q fragments held in registers for the whole sweep (4 x 32-deep chunks)
    v16h qf[4];
    const _Float16* qrow = Qh + (size_t)(m0 + m) * DH;
#pragma unroll
    for (int kk = 0; kk < 4; ++kk) qf[kk] = load_frag(qrow + kk * 32, hi);

    v8f zero = {};
    v8f acc[8];
#pragma unroll
    for (int i = 0; i < 8; ++i) acc[i] = zero;
    float mrow[8], lrow[8];
#pragma unroll
    for (int r = 0; r < 8; ++r) { mrow[r] = -1e30f; lrow[r] = 0.0f; }

    for (int k0 = kstart; k0 < kend; k0 += 32) {
        if (k0 + 32 < kend) {  // hint next K/V block toward the caches
            __builtin_prefetch(Kh + (size_t)(k0 + 32 + m) * DH, 0, 1);
            __builtin_prefetch(Vt + (size_t)m * NTOK + k0 + 32, 0, 1);
        }

        // S = Q * K^T for 32 keys -> two 16x16 f32 tiles
        v8f s[2];
#pragma unroll
        for (int t = 0; t < 2; ++t) {
            v8f cs = zero;
            const _Float16* krow = Kh + (size_t)(k0 + t * 16 + m) * DH;
#pragma unroll
            for (int kk = 0; kk < 4; ++kk) {
                v16h bf = load_frag(krow + kk * 32, hi);
                cs = __builtin_amdgcn_wmma_f32_16x16x32_f16(false, qf[kk], false, bf,
                                                            (short)0, cs, false, false);
            }
            s[t] = cs;
        }

        // online softmax (per C-layout row r+8*hi; stats shared across 16 lanes)
#pragma unroll
        for (int r = 0; r < 8; ++r) {
            float s0 = s[0][r], s1 = s[1][r];
            float tm = red_max16(fmaxf(s0, s1));
            float mn = fmaxf(mrow[r], tm);
            float alpha = __expf(mrow[r] - mn);
            float p0 = __expf(s0 - mn);
            float p1 = __expf(s1 - mn);
            float rs = red_sum16(p0 + p1);
            lrow[r] = lrow[r] * alpha + rs;
            mrow[r] = mn;
#pragma unroll
            for (int d = 0; d < 8; ++d) acc[d][r] *= alpha;
            // stage P (row-major [16,32] f16) for the A-fragment reload
            pl[(r + 8 * hi) * 32 + m]      = (_Float16)p0;
            pl[(r + 8 * hi) * 32 + 16 + m] = (_Float16)p1;
        }

        // make the cross-lane LDS writes visible before fragment reload
        asm volatile("s_wait_dscnt 0" ::: "memory");
        v16h pf = load_frag(pl + m * 32, hi);

        // O += P * V   (B^T rows = Vt[dim][token], contiguous along tokens)
#pragma unroll
        for (int d = 0; d < 8; ++d) {
            const _Float16* vrow = Vt + (size_t)(d * 16 + m) * NTOK + k0;
            v16h vf = load_frag(vrow, hi);
            acc[d] = __builtin_amdgcn_wmma_f32_16x16x32_f16(false, pf, false, vf,
                                                            (short)0, acc[d], false, false);
        }
        asm volatile("" ::: "memory");   // keep iter i reads before iter i+1 LDS writes
    }

    // epilogue: store unnormalized partials + per-row stats
#pragma unroll
    for (int r = 0; r < 8; ++r) {
        int gm = m0 + r + 8 * hi;
#pragma unroll
        for (int d = 0; d < 8; ++d)
            Opart[((size_t)seg * NTOK + gm) * DH + d * 16 + m] = acc[d][r];
        if (m == 0) {                    // one lane per row writes stats
            Mpart[seg * NTOK + gm] = mrow[r];
            Lpart[seg * NTOK + gm] = lrow[r];
        }
    }
}

// ---------------- k4: combine segment partials ----------------
__global__ void attn_combine_kernel(const float* __restrict__ Opart,
                                    const float* __restrict__ Mpart,
                                    const float* __restrict__ Lpart,
                                    float* __restrict__ out) {
    int i = blockIdx.x * blockDim.x + threadIdx.x;
    if (i >= NTOK * DH) return;
    int row = i >> 7;                    // /DH
    float M = -1e30f;
#pragma unroll
    for (int s = 0; s < NSEG; ++s) M = fmaxf(M, Mpart[s * NTOK + row]);
    float L = 0.0f, o = 0.0f;
#pragma unroll
    for (int s = 0; s < NSEG; ++s) {
        float w = __expf(Mpart[s * NTOK + row] - M);
        L += w * Lpart[s * NTOK + row];
        o += w * Opart[(size_t)s * NTOK * DH + i];
    }
    out[i] = o / L;
}

extern "C" void kernel_launch(void* const* d_in, const int* in_sizes, int n_in,
                              void* d_out, int out_size, void* d_ws, size_t ws_size,
                              hipStream_t stream) {
    const float* x  = (const float*)d_in[0];
    const float* wq = (const float*)d_in[1];
    const float* wk = (const float*)d_in[2];
    const float* wv = (const float*)d_in[3];
    float* out = (float*)d_out;

    // workspace layout (~27 MB total; partials stay resident in 192 MB L2)
    char* ws = (char*)d_ws;
    size_t o = 0;
    _Float16* xh    = (_Float16*)(ws + o); o += (size_t)NTOK * DIN * 2;        // 4 MB
    _Float16* wt    = (_Float16*)(ws + o); o += (size_t)3 * DH * DIN * 2;      // 192 KB
    _Float16* Qh    = (_Float16*)(ws + o); o += (size_t)NTOK * DH * 2;         // 2 MB
    _Float16* Kh    = (_Float16*)(ws + o); o += (size_t)NTOK * DH * 2;         // 2 MB
    _Float16* Vt    = (_Float16*)(ws + o); o += (size_t)DH * NTOK * 2;         // 2 MB
    float*    Opart = (float*)(ws + o);    o += (size_t)NSEG * NTOK * DH * 4;  // 16 MB
    float*    Mpart = (float*)(ws + o);    o += (size_t)NSEG * NTOK * 4;       // 128 KB
    float*    Lpart = (float*)(ws + o);    o += (size_t)NSEG * NTOK * 4;       // 128 KB
    (void)o; (void)ws_size; (void)in_sizes; (void)n_in; (void)out_size;

    // k0: x -> f16
    {
        int n = NTOK * DIN;
        cvt_x_kernel<<<(n + 255) / 256, 256, 0, stream>>>(x, xh, n);
    }
    // k1: weights -> transposed f16
    {
        int n = 3 * DH * DIN;
        cvt_w_kernel<<<(n + 255) / 256, 256, 0, stream>>>(wq, wk, wv, wt);
    }
    // k2: QKV GEMM. 3 mats * 512 * 8 tiles = 12288 waves; 8 waves/block
    qkv_gemm_kernel<<<12288 / 8, 256, 0, stream>>>(xh, wt, Qh, Kh, Vt);
    // k3: split-K flash attention. 512 tiles * 4 segments = 2048 waves; 4/block
    flash_attn_partial_kernel<<<2048 / 4, 128, 0, stream>>>(Qh, Kh, Vt,
                                                            Opart, Mpart, Lpart);
    // k4: combine. one thread per output element
    attn_combine_kernel<<<(NTOK * DH + 255) / 256, 256, 0, stream>>>(Opart, Mpart,
                                                                     Lpart, out);
}
